// ExRNN_41145786696082
// MI455X (gfx1250) — compile-verified, matrix operand order checked by reading, not run on hardware
//
#include <hip/hip_runtime.h>

typedef __attribute__((ext_vector_type(16))) _Float16 v16h;
typedef __attribute__((ext_vector_type(8)))  _Float16 v8h;
typedef __attribute__((ext_vector_type(8)))  float    v8f;
typedef __attribute__((ext_vector_type(4)))  float    v4f;

#define T_LEN 4096
#define B_SZ  256
#define D_IN  64
#define H_SZ  128
#define O_SZ  64
#define K_ALL (D_IN + H_SZ)   // 192: fused [x | h] contraction, matches W_in column layout

#define HPAD 136   // halves per padded h row: 272B, 16B-aligned, bank-skewed
#define XPAD 72    // halves per padded x row: 144B, 16B-aligned
#define BLOCK 384  // 12 waves: 8 h-update waves + 4 y/x-stage waves

// Branch-free tanh: 1 - 2/(exp(2x)+1), via v_exp_f32 + v_rcp_f32.
// Saturates correctly: x->+inf => exp2->inf => rcp->0 => 1;  x->-inf => exp2->0 => -1.
__device__ __forceinline__ float fast_tanh(float x) {
  const float e = __builtin_amdgcn_exp2f(x * 2.885390081777927f);  // exp(2x)
  return 1.0f - 2.0f * __builtin_amdgcn_rcpf(e + 1.0f);
}

// A-fragment (16x32 f16, row-per-lane) from an LDS tile.
// lane<16: row=nl, K = kbase..kbase+7 and kbase+16..kbase+23 (kbase = kk*32)
// lane>=16: same rows, K shifted by +8 (kbase = kk*32 + 8)
__device__ __forceinline__ v16h frag_from_lds(const _Float16* __restrict__ base,
                                              int row, int rowstride, int kbase) {
  const v8h lo = *(const v8h*)(base + row * rowstride + kbase);
  const v8h hi = *(const v8h*)(base + row * rowstride + kbase + 16);
  return __builtin_shufflevector(lo, hi, 0,1,2,3,4,5,6,7,8,9,10,11,12,13,14,15);
}

// B-fragment (32x16 f16, column-per-lane) from a global f32 weight row (B[k][n] = W[n][k]).
__device__ __forceinline__ v16h bfrag_from_global(const float* __restrict__ row, int kbase) {
  v16h b;
  #pragma unroll
  for (int e = 0; e < 8; ++e) {
    b[e]     = (_Float16)row[kbase + e];
    b[8 + e] = (_Float16)row[kbase + 16 + e];
  }
  return b;
}

__global__ __launch_bounds__(BLOCK, 1)
void exrnn_fused_scan(const float* __restrict__ x,      // (T,B,D)
                      const float* __restrict__ h0,     // (B,H)
                      const float* __restrict__ Win,    // (H, D+H)
                      const float* __restrict__ bin,    // (H)
                      const float* __restrict__ Wout,   // (O, H)
                      const float* __restrict__ bout,   // (O)
                      float* __restrict__ out)          // (B,T,O) then (B,H)
{
  __shared__ _Float16 hbuf[2][16 * HPAD];    // double-buffered hidden state (f16)
  __shared__ _Float16 xstage[2][16 * XPAD];  // double-buffered x_t tile (f16)

  const int tid   = threadIdx.x;
  const int wave  = tid >> 5;
  const int lane  = tid & 31;
  const int nl    = lane & 15;
  const int hi    = lane >> 4;       // half-wave select (K/row split per ISA layout)
  const int brow0 = blockIdx.x * 16; // 16 batch rows per workgroup

  // ---- initial hidden state -> hbuf[0] ----
  for (int i = tid; i < 16 * H_SZ; i += BLOCK) {
    const int r = i >> 7, c = i & 127;
    hbuf[0][r * HPAD + c] = (_Float16)h0[(size_t)(brow0 + r) * H_SZ + c];
  }

  // ---------------- per-wave persistent state ----------------
  v16h  WF[6];          // h-waves: W_in fragments (K=192 -> 6 k-steps)
  v16h  WOF[4];         // y-waves: W_out fragments (K=128 -> 4 k-steps)
  float bias = 0.f, ybias = 0.f;
  float hfin[8];
  int   col = 0, ycol = 0, srow = 0, scol = 0;
  v4f   xc0 = {}, xc1 = {};   // pending x loads (for step t+1)

  if (wave < 8) {
    __builtin_amdgcn_s_setprio(2);      // h-update chain is the critical path
    col  = wave * 16 + nl;              // H column owned by this lane
    bias = bin[col];
    const float* wrow = Win + (size_t)col * K_ALL;
    #pragma unroll
    for (int kk = 0; kk < 6; ++kk)
      WF[kk] = bfrag_from_global(wrow, kk * 32 + hi * 8);
    #pragma unroll
    for (int r = 0; r < 8; ++r) hfin[r] = 0.f;
  } else {
    __builtin_amdgcn_s_setprio(0);      // y / staging waves run one step behind
    ycol  = (wave - 8) * 16 + nl;       // O column owned by this lane
    ybias = bout[ycol];
    const float* wrow = Wout + (size_t)ycol * H_SZ;
    #pragma unroll
    for (int kk = 0; kk < 4; ++kk)
      WOF[kk] = bfrag_from_global(wrow, kk * 32 + hi * 8);

    // x staging: 128 lanes cover the 16x64 f32 tile, 8 floats each
    const int L = tid - 256;            // 0..127
    srow = L >> 3;
    scol = (L & 7) * 8;
    {   // stage x_0 now; leave x_1 pending in registers
      const float* p = x + ((size_t)0 * B_SZ + brow0 + srow) * D_IN + scol;
      const v4f a0 = *(const v4f*)p;
      const v4f a1 = *(const v4f*)(p + 4);
      v8h s;
      #pragma unroll
      for (int e = 0; e < 4; ++e) { s[e] = (_Float16)a0[e]; s[4 + e] = (_Float16)a1[e]; }
      *(v8h*)(xstage[0] + srow * XPAD + scol) = s;
      const float* q = x + ((size_t)1 * B_SZ + brow0 + srow) * D_IN + scol;
      xc0 = *(const v4f*)q;
      xc1 = *(const v4f*)(q + 4);
    }
  }

  // ---------------- sequential scan: T+1 iterations, one barrier each ----------------
  #pragma unroll 2
  for (int t = 0; t <= T_LEN; ++t) {
    __syncthreads();

    if (wave < 8) {
      if (t < T_LEN) {
        const _Float16* xs = xstage[t & 1];
        const _Float16* hb = hbuf[t & 1];
        _Float16*       hn = hbuf[(t & 1) ^ 1];

        // Hoist ALL A-fragment LDS loads before the WMMA chain so the compiler
        // can use graduated s_wait_dscnt values instead of dscnt==0 per WMMA.
        v16h afrag[6];
        #pragma unroll
        for (int kk = 0; kk < 2; ++kk)
          afrag[kk] = frag_from_lds(xs, nl, XPAD, kk * 32 + hi * 8);
        #pragma unroll
        for (int kk = 0; kk < 4; ++kk)
          afrag[2 + kk] = frag_from_lds(hb, nl, HPAD, kk * 32 + hi * 8);

        // Two accumulator chains, both starting from the inline-0 C operand
        // (no splat movs); per-lane bias is added once at the end.
        v8f acc  = {};
        v8f acc2 = {};
        #pragma unroll
        for (int kk = 0; kk < 6; ++kk) {
          if (kk & 1)
            acc2 = __builtin_amdgcn_wmma_f32_16x16x32_f16(false, afrag[kk], false, WF[kk],
                                                          (short)0, acc2, false, false);
          else
            acc  = __builtin_amdgcn_wmma_f32_16x16x32_f16(false, afrag[kk], false, WF[kk],
                                                          (short)0, acc, false, false);
        }

        #pragma unroll
        for (int r = 0; r < 8; ++r) {
          const float hv = fast_tanh(acc[r] + acc2[r] + bias);  // branch-free, EXEC stays full
          hfin[r] = hv;
          hn[(r + hi * 8) * HPAD + col] = (_Float16)hv;   // C layout: VGPR r -> M = r + hi*8
        }
      } else {
        // t == T: emit h_final (f32 accumulators from the last step)
        float* hf = out + (size_t)B_SZ * T_LEN * O_SZ;
        #pragma unroll
        for (int r = 0; r < 8; ++r)
          hf[(size_t)(brow0 + r + hi * 8) * H_SZ + col] = hfin[r];
      }
    } else {
      // ---- y / x-staging waves ----
      v4f xn0 = xc0, xn1 = xc1;
      if (t + 2 < T_LEN) {             // issue loads for x_{t+2} (2-step prefetch)
        const float* p = x + ((size_t)(t + 2) * B_SZ + brow0 + srow) * D_IN + scol;
        xn0 = *(const v4f*)p;
        xn1 = *(const v4f*)(p + 4);
      }
      if (t + 1 < T_LEN) {             // stage x_{t+1} into the buffer the h-waves read next
        v8h s;
        #pragma unroll
        for (int e = 0; e < 4; ++e) { s[e] = (_Float16)xc0[e]; s[4 + e] = (_Float16)xc1[e]; }
        *(v8h*)(xstage[(t + 1) & 1] + srow * XPAD + scol) = s;
      }
      xc0 = xn0; xc1 = xn1;

      if (t > 0) {                     // y_{t-1} from h'_{t-1} (in hbuf[t&1])
        const _Float16* hb = hbuf[t & 1];

        v16h afrag[4];
        #pragma unroll
        for (int kk = 0; kk < 4; ++kk)
          afrag[kk] = frag_from_lds(hb, nl, HPAD, kk * 32 + hi * 8);

        v8f ya  = {};
        v8f ya2 = {};
        #pragma unroll
        for (int kk = 0; kk < 4; ++kk) {
          if (kk & 1)
            ya2 = __builtin_amdgcn_wmma_f32_16x16x32_f16(false, afrag[kk], false, WOF[kk],
                                                         (short)0, ya2, false, false);
          else
            ya  = __builtin_amdgcn_wmma_f32_16x16x32_f16(false, afrag[kk], false, WOF[kk],
                                                         (short)0, ya, false, false);
        }

        #pragma unroll
        for (int r = 0; r < 8; ++r)
          out[((size_t)(brow0 + r + hi * 8) * T_LEN + (t - 1)) * O_SZ + ycol] =
              ya[r] + ya2[r] + ybias;
      }
    }
  }
}

extern "C" void kernel_launch(void* const* d_in, const int* in_sizes, int n_in,
                              void* d_out, int out_size, void* d_ws, size_t ws_size,
                              hipStream_t stream) {
  (void)in_sizes; (void)n_in; (void)out_size; (void)d_ws; (void)ws_size;
  const float* x    = (const float*)d_in[0];
  const float* h0   = (const float*)d_in[1];
  const float* Win  = (const float*)d_in[2];
  const float* bin  = (const float*)d_in[3];
  const float* Wout = (const float*)d_in[4];
  const float* bout = (const float*)d_in[5];
  float* out = (float*)d_out;

  dim3 grid(B_SZ / 16);   // 16 workgroups, one 16-row batch tile each
  dim3 block(BLOCK);      // 12 waves
  exrnn_fused_scan<<<grid, block, 0, stream>>>(x, h0, Win, bin, Wout, bout, out);
}
